// VanillaRNN_14731737825790
// MI455X (gfx1250) — matrix operand order repaired and checked by reading, not compiled, not run on hardware
//
#include <hip/hip_runtime.h>

// Vanilla RNN forward for MI455X (gfx1250, wave32, WMMA + TDM).
// T=1024 steps, B=128, H=2048, VOCAB=128. Persistent kernel, 1 grid sync/step.
//
//  - 128 "W" blocks: block owns batch-tile cb0=(blk&7)*16 and 8 row-tiles
//    (wave w -> rows ((blk>>3)*8+w)*16). Per step the block stages its 16 h
//    columns (16 x 2056 bf16, padded stride) into LDS with ONE
//    TENSOR_LOAD_TO_LDS (TDM, s_wait_tensorcnt), then each wave runs
//    64 x v_wmma_f32_16x16x32_bf16 with A from global (L2) and B from LDS
//    (ds_load_b128, conflict-mitigated by the 2056-element row stride).
//  - 8 "V" blocks: y_{t-1} = V@h_t + by concurrently with the next W@h.
//
// Layouts: Wb/Vb bf16 row-major over K (A-frag = 2 contiguous 16B runs/lane);
// hT bf16 transposed (B x 2056), K contiguous (B-frag = 2 contiguous 16B runs);
// h32 fp32 (H,B) side copy -> exact final h_T output.

#define T_STEPS 1024
#define BATCH   128
#define HID     2048
#define HSTR    2056   // padded hT row stride (elements): 1028 dwords == 4 mod 64 banks
#define VOC     128
#define NWBLK   128
#define NVBLK   8
#define NBLK    (NWBLK + NVBLK)
#define KSTEPS  (HID / 32)

typedef __attribute__((ext_vector_type(16))) __bf16 v16bf;
typedef __attribute__((ext_vector_type(8)))  float  v8f;
typedef __attribute__((ext_vector_type(4)))  unsigned int v4u;
typedef __attribute__((ext_vector_type(8)))  int    v8i;
typedef __attribute__((ext_vector_type(4)))  int    v4i;

union FragAB { v16bf v; uint4 q[2]; };
union Half8  { uint4 q; unsigned short s[8]; };

__device__ __forceinline__ unsigned short f32_to_bf16(float f) {
  unsigned int u = __float_as_uint(f);
  u += 0x7FFFu + ((u >> 16) & 1u);   // round-to-nearest-even
  return (unsigned short)(u >> 16);
}

// branchless tanh: inputs are bounded in this net; exact enough vs bf16 state
__device__ __forceinline__ float fast_tanh(float x) {
  float xc = fminf(fmaxf(x, -15.f), 15.f);
  float e  = __expf(2.f * xc);
  return 1.f - 2.f / (e + 1.f);
}

// ---------------- prep: bf16 copies of W, V; init hT (padded) and h32 ----------------
__global__ void rnn_prep(const float* __restrict__ W, const float* __restrict__ V,
                         const float* __restrict__ h0,
                         unsigned short* __restrict__ Wb, unsigned short* __restrict__ Vb,
                         unsigned short* __restrict__ hTa, float* __restrict__ h32) {
  size_t tid    = (size_t)blockIdx.x * blockDim.x + threadIdx.x;
  size_t stride = (size_t)gridDim.x * blockDim.x;
  for (size_t i = tid; i < (size_t)HID * HID; i += stride) Wb[i] = f32_to_bf16(W[i]);
  for (size_t i = tid; i < (size_t)VOC * HID; i += stride) Vb[i] = f32_to_bf16(V[i]);
  for (size_t i = tid; i < (size_t)HID * BATCH; i += stride) {
    size_t b = i / HID, k = i % HID;          // hTa[b*HSTR + k] = h0[k*BATCH + b]
    hTa[b * HSTR + k] = f32_to_bf16(h0[k * BATCH + b]);
  }
  for (size_t i = tid; i < (size_t)HID * BATCH; i += stride) h32[i] = h0[i];
}

// K-loop: A row-fragments from global, B column-fragments from LDS
__device__ __forceinline__ v8f mm_tile(const unsigned short* __restrict__ Arow,
                                       const char* __restrict__ bRow, int g) {
  v8f c = {};
  const int ga = g * 8;         // A: element offset of first 8-elem run
  const int gboff = g * 16;     // B: byte offset of first 16B run
  #pragma unroll 4
  for (int kt = 0; kt < KSTEPS; ++kt) {
    const int kb = kt * 32;
    FragAB fa, fb;
    fa.q[0] = *(const uint4*)(Arow + kb + ga);          // k = kb+g*8    .. +8
    fa.q[1] = *(const uint4*)(Arow + kb + 16 + ga);     // k = kb+16+g*8 .. +8
    fb.q[0] = *(const uint4*)(bRow + kb * 2 + gboff);        // k = kb+g*8
    fb.q[1] = *(const uint4*)(bRow + kb * 2 + gboff + 32);   // k = kb+16+g*8
    c = __builtin_amdgcn_wmma_f32_16x16x32_bf16(false, fa.v, false, fb.v,
                                                (short)0, c, false, false);
  }
  return c;
}

// ---------------- persistent RNN kernel ----------------
__global__ __launch_bounds__(256) void rnn_persistent(
    const int*   __restrict__ tok,   // (T,B)
    const float* __restrict__ U,     // (H,VOC)
    const float* __restrict__ bh,    // (H)
    const float* __restrict__ by,    // (VOC)
    const unsigned short* __restrict__ Wb,   // (H,H) bf16
    const unsigned short* __restrict__ Vb,   // (VOC,H) bf16
    unsigned short* hTa, unsigned short* hTb,  // (B,HSTR) bf16 ping/pong
    float* __restrict__ h32,                   // (H,B) fp32 current state
    float* __restrict__ out,                   // logits (T,B,VOC) ++ h_T (H,B)
    unsigned int* __restrict__ cnt)            // grid barrier counter (memset 0)
{
  __shared__ unsigned short sB[16 * HSTR];     // staged h columns (TDM dest)

  const int lane = threadIdx.x & 31;
  const int wave = threadIdx.x >> 5;
  const int blk  = blockIdx.x;
  const int g    = lane >> 4;   // half-wave group (0/1)
  const int hl   = lane & 15;   // 0..15
  const bool isW = (blk < NWBLK);

  // both kinds of block own one 16-wide batch tile; waves own 16-row tiles
  const int cb0 = isW ? (blk & 7) * 16 : (blk - NWBLK) * 16;       // batch cols
  const int m0  = isW ? ((blk >> 3) * 8 + wave) * 16 : wave * 16;  // out rows

  const char* bRow = (const char*)sB + (size_t)hl * (HSTR * 2);    // lane's B column

  unsigned short* hcur = hTa;   // state h_t (bf16, transposed, padded stride)
  unsigned short* hnxt = hTb;

  for (int t = 0; t <= T_STEPS; ++t) {
    const bool active = isW ? (t < T_STEPS) : (t >= 1);
    if (active) {
      // ---- stage hcur[cb0 : cb0+16, :] into LDS with one TDM op ----
      if (wave == 0) {
        unsigned long long ga = (unsigned long long)(size_t)(hcur + (size_t)cb0 * HSTR);
        v4u g0;
        g0.x = 1u;                                  // count=1, user mode, no gather
        g0.y = (unsigned)(size_t)sB;                // lds_addr (low 32 bits = LDS offset)
        g0.z = (unsigned)(ga & 0xFFFFFFFFull);      // global_addr[31:0]
        g0.w = (unsigned)((ga >> 32) & 0x01FFFFFFull) | 0x80000000u;  // addr[56:32], type=2
        v8i g1;
        g1[0] = 0x00010000;            // workgroup_mask=0, data_size=1 (2 bytes)
        g1[1] = (int)(HSTR << 16);     // tensor_dim0 = 2056 (low 16 in [63:48])
        g1[2] = (int)(BATCH << 16);    // dim0 hi=0, tensor_dim1 = 128
        g1[3] = (int)(HSTR << 16);     // dim1 hi=0, tile_dim0 = 2056
        g1[4] = 16;                    // tile_dim1 = 16, tile_dim2 = 0
        g1[5] = HSTR;                  // tensor_dim0_stride = 2056 (low 32)
        g1[6] = 0;                     // stride hi = 0, tensor_dim1_stride = 0
        g1[7] = 0;
        v4i z4 = {0, 0, 0, 0};
#if __clang_major__ >= 23
        v8i z8 = {0, 0, 0, 0, 0, 0, 0, 0};
        __builtin_amdgcn_tensor_load_to_lds(g0, g1, z4, z4, z8, 0);
#else
        __builtin_amdgcn_tensor_load_to_lds(g0, g1, z4, z4, 0);
#endif
        __builtin_amdgcn_s_wait_tensorcnt((short)0);
      }
      __syncthreads();

      if (isW) {
        // ---- hnew[m0:+16, cb0:+16] = tanh(Ux_t + W@h + bh) ----
        v8f c = mm_tile(Wb + (size_t)(m0 + hl) * HID, bRow, g);
        const int token = tok[(size_t)t * BATCH + cb0 + hl];
        Half8 hb;
        #pragma unroll
        for (int v = 0; v < 8; ++v) {
          const int m = m0 + g * 8 + v;                    // C/D: row M = 8g+v, col N = hl
          float val = c[v] + U[(size_t)m * VOC + token] + bh[m];
          val = fast_tanh(val);
          h32[(size_t)m * BATCH + cb0 + hl] = val;         // fp32 state (final h_T source)
          hb.s[v] = f32_to_bf16(val);
        }
        *(uint4*)(hnxt + (size_t)(cb0 + hl) * HSTR + m0 + g * 8) = hb.q;  // one 16B store
      } else {
        // ---- y_{t-1} = V @ h_t + by ; logits[t-1][b][o] ----
        v8f c = mm_tile(Vb + (size_t)(m0 + hl) * HID, bRow, g);
        float* o = out + (size_t)(t - 1) * (BATCH * VOC)
                       + (size_t)(cb0 + hl) * VOC + m0 + g * 8;  // 32B contiguous per lane
        float4 r0, r1;
        r0.x = c[0] + by[m0 + g * 8 + 0];
        r0.y = c[1] + by[m0 + g * 8 + 1];
        r0.z = c[2] + by[m0 + g * 8 + 2];
        r0.w = c[3] + by[m0 + g * 8 + 3];
        r1.x = c[4] + by[m0 + g * 8 + 4];
        r1.y = c[5] + by[m0 + g * 8 + 5];
        r1.z = c[6] + by[m0 + g * 8 + 6];
        r1.w = c[7] + by[m0 + g * 8 + 7];
        *(float4*)(o)     = r0;
        *(float4*)(o + 4) = r1;
      }
    }

    // ---- grid barrier (monotonic counter; reset by memset each launch) ----
    __threadfence();
    __syncthreads();
    if (threadIdx.x == 0) {
      __hip_atomic_fetch_add(cnt, 1u, __ATOMIC_ACQ_REL, __HIP_MEMORY_SCOPE_AGENT);
      const unsigned int target = (unsigned int)(t + 1) * (unsigned int)gridDim.x;
      while (__hip_atomic_load(cnt, __ATOMIC_ACQUIRE, __HIP_MEMORY_SCOPE_AGENT) < target)
        __builtin_amdgcn_s_sleep(2);
    }
    __syncthreads();
    __threadfence();

    unsigned short* tmp = hcur; hcur = hnxt; hnxt = tmp;
  }

  // ---- append h_T (fp32) after logits ----
  for (size_t i = (size_t)blk * blockDim.x + threadIdx.x; i < (size_t)HID * BATCH;
       i += (size_t)gridDim.x * blockDim.x)
    out[(size_t)T_STEPS * BATCH * VOC + i] = h32[i];
}

extern "C" void kernel_launch(void* const* d_in, const int* in_sizes, int n_in,
                              void* d_out, int out_size, void* d_ws, size_t ws_size,
                              hipStream_t stream) {
  const int*   tok = (const int*)  d_in[0];   // (T,B) int32
  const float* h0  = (const float*)d_in[1];   // (H,B)
  const float* U   = (const float*)d_in[2];   // (H,VOC)
  const float* W   = (const float*)d_in[3];   // (H,H)
  const float* V   = (const float*)d_in[4];   // (VOC,H)
  const float* bh  = (const float*)d_in[5];   // (H,1)
  const float* by  = (const float*)d_in[6];   // (VOC,1)
  float* out = (float*)d_out;

  // workspace carve-up (~10.6 MB)
  char* p = (char*)d_ws;
  unsigned short* Wb  = (unsigned short*)p; p += (size_t)HID * HID * 2;     // 8 MB
  unsigned short* Vb  = (unsigned short*)p; p += (size_t)VOC * HID * 2;     // 512 KB
  unsigned short* hTa = (unsigned short*)p; p += (size_t)BATCH * HSTR * 2;  // ~514 KB
  unsigned short* hTb = (unsigned short*)p; p += (size_t)BATCH * HSTR * 2;  // ~514 KB
  float*          h32 = (float*)p;          p += (size_t)HID * BATCH * 4;   // 1 MB
  unsigned int*   cnt = (unsigned int*)p;

  hipMemsetAsync(cnt, 0, 64, stream);                       // barrier counter = 0
  rnn_prep<<<512, 256, 0, stream>>>(W, V, h0, Wb, Vb, hTa, h32);
  rnn_persistent<<<NBLK, 256, 0, stream>>>(tok, U, bh, by, Wb, Vb,
                                           hTa, hTb, h32, out, cnt);
}
